// GroupedQueryAttention_57406532879058
// MI455X (gfx1250) — matrix operand order repaired and checked by reading, not compiled, not run on hardware
//
#include <hip/hip_runtime.h>

typedef __attribute__((ext_vector_type(16))) __bf16 v16bf;
typedef __attribute__((ext_vector_type(8)))  float  v8f;

#define B_   2
#define L_   2048
#define D_   2048
#define NH_  16
#define NKV_ 8
#define HD_  128
#define SCALE_ 0.08838834764831845f  /* 128^-0.5 */

// global -> LDS async copy of 16 bytes per lane (ASYNCcnt-tracked, no VGPR data)
__device__ __forceinline__ void async_b128(const void* gptr, const void* lptr) {
  unsigned lds = (unsigned)(size_t)lptr;               // flat addr[31:0] == LDS offset
  unsigned long long g = (unsigned long long)(size_t)gptr;
  asm volatile("global_load_async_to_lds_b128 %0, %1, off"
               :: "v"(lds), "v"(g) : "memory");
}
__device__ __forceinline__ void wait_async0() {
  asm volatile("s_wait_asynccnt 0" ::: "memory");
}

// ---------------------------------------------------------------- conversion
__global__ void cvt_f32_bf16(const float* __restrict__ in, __bf16* __restrict__ out, int n) {
  int i = blockIdx.x * blockDim.x + threadIdx.x;
  if (i < n) out[i] = (__bf16)in[i];
}

// ---------------------------------------------------------------- bf16 GEMM
// C[M,N] = A[M,K] * W[N,K]^T  (A,W bf16 row-major; C = OutT row-major)
// 256 threads = 8 waves; block tile 64x128; wave tile 32x32 (2x2 WMMA).
// A tiles stream via async global->LDS b128, double buffered.
template <typename OutT>
__global__ __launch_bounds__(256)
void gemm_bf16(const __bf16* __restrict__ A, const __bf16* __restrict__ W,
               OutT* __restrict__ C, int M, int N, int K) {
  __shared__ __bf16 As[2][64][48];    // rows 96B -> all b128 dsts 16B aligned
  __shared__ __bf16 Bt[2][32][136];   // [k][n], transposed at store
  const int tid  = threadIdx.x;
  const int lane = tid & 31, w = tid >> 5;
  const int wm = w >> 2, wn = w & 3;
  const int half = lane >> 4, l15 = lane & 15;
  const int mbase = blockIdx.y * 64, nbase = blockIdx.x * 128;

  // one async b128 per thread covers the whole 64x32 A tile
  const int ar = tid >> 2;            // 0..63
  const int ac = (tid & 3) * 8;       // 0,8,16,24
  const __bf16* aptr = A + (size_t)(mbase + ar) * K + ac;

  v8f acc[2][2];
  for (int i = 0; i < 2; i++)
    for (int j = 0; j < 2; j++)
      for (int e = 0; e < 8; e++) acc[i][j][e] = 0.0f;

  auto issueA = [&](int buf, int k0) {
    async_b128(aptr + k0, &As[buf][ar][ac]);
  };
  auto loadB = [&](int buf, int k0) {
    for (int i = tid; i < 128 * 32; i += 256) {
      int n = i >> 5, kk = i & 31;
      Bt[buf][kk][n] = W[(size_t)(nbase + n) * K + k0 + kk];
    }
  };

  issueA(0, 0);
  loadB(0, 0);
  wait_async0();
  __syncthreads();

  for (int k0 = 0; k0 < K; k0 += 32) {
    const int cur = (k0 >> 5) & 1;
    if (k0 + 32 < K) {                 // stream next tile while computing
      issueA(cur ^ 1, k0 + 32);
      loadB(cur ^ 1, k0 + 32);
      __builtin_prefetch(W + (size_t)(nbase + (tid >> 1)) * K + k0 + 64, 0, 1);
    }

    v16bf a[2], b[2];
    for (int i = 0; i < 2; i++) {
      int r = wm * 32 + i * 16 + l15;
      for (int e = 0; e < 8; e++) {
        a[i][e]     = As[cur][r][half * 8 + e];
        a[i][8 + e] = As[cur][r][16 + half * 8 + e];
      }
    }
    for (int j = 0; j < 2; j++) {
      int c0 = wn * 32 + j * 16;
      for (int e = 0; e < 8; e++) {
        b[j][e]     = Bt[cur][half * 8 + e][c0 + l15];
        b[j][8 + e] = Bt[cur][16 + half * 8 + e][c0 + l15];
      }
    }
    for (int i = 0; i < 2; i++)
      for (int j = 0; j < 2; j++)
        acc[i][j] = __builtin_amdgcn_wmma_f32_16x16x32_bf16(
            false, a[i], false, b[j], (short)0, acc[i][j], false, false);

    wait_async0();
    __syncthreads();
  }

  for (int i = 0; i < 2; i++)
    for (int j = 0; j < 2; j++)
      for (int v = 0; v < 8; v++) {
        int row = mbase + wm * 32 + i * 16 + v + 8 * half;
        int col = nbase + wn * 32 + j * 16 + l15;
        C[(size_t)row * N + col] = (OutT)acc[i][j][v];
      }
}

// ---------------------------------------------------------------- RoPE+split
__global__ void rope_split(const __bf16* __restrict__ qkv,
                           __bf16* __restrict__ Qr, __bf16* __restrict__ Kr,
                           __bf16* __restrict__ Vr) {
  int idx = blockIdx.x * 256 + threadIdx.x;
  int row = idx >> 12;
  int n   = idx & 4095;
  int b   = row >> 11;
  int pos = row & 2047;
  __bf16 raw = qkv[idx];
  if (n < 3072) {
    int base = (n < 2048) ? 0 : 2048;
    int m = n - base;
    int h = m >> 7, d = m & 127;
    int i = d & 63;
    float inv = __expf(-((float)(2 * i) * (1.0f / 128.0f)) * 9.210340371976184f);
    float fr = (float)pos * inv, s, c;
    __sincosf(fr, &s, &c);
    int pidx = (row << 12) + (n - d) + ((d < 64) ? d + 64 : d - 64);
    float pv  = (float)qkv[pidx];
    float rot = (d < 64) ? -pv : pv;
    float out = (float)raw * c + rot * s;
    if (n < 2048)
      Qr[(((size_t)b * NH_ + h) * L_ + pos) * HD_ + d] = (__bf16)out;
    else
      Kr[(((size_t)b * NKV_ + h) * L_ + pos) * HD_ + d] = (__bf16)out;
  } else {
    int m = n - 3072, h = m >> 7, d = m & 127;
    Vr[(((size_t)b * NKV_ + h) * L_ + pos) * HD_ + d] = raw;
  }
}

// ---------------------------------------------------------------- flash attn
// 128 threads = 4 waves; wave owns 16 query rows; key tiles of 32, K/V double
// buffered; Q and V staged via async global->LDS; softmax online in f32.
__global__ __launch_bounds__(128)
void flash_gqa(const __bf16* __restrict__ Qr, const __bf16* __restrict__ Kr,
               const __bf16* __restrict__ Vr, __bf16* __restrict__ Oa) {
  __shared__ __bf16 Qs[64][136];      // [qrow][hd], rows 272B (16B aligned)
  __shared__ __bf16 Kt[2][128][40];   // [hd][key]
  __shared__ __bf16 Vs[2][32][136];   // [key][hd]
  __shared__ __bf16 Ps[4][16][40];    // per-wave P bounce

  int bid = blockIdx.x;
  int qt  = bid & 31;
  int h   = (bid >> 5) & 15;
  int b   = bid >> 9;
  int kvh = h >> 1;
  int tid = threadIdx.x, lane = tid & 31, w = tid >> 5;
  int half = lane >> 4, l15 = lane & 15;

  const __bf16* Qbase = Qr + (((size_t)b * NH_ + h) * L_ + qt * 64) * HD_;
  const __bf16* Kbase = Kr + ((size_t)b * NKV_ + kvh) * L_ * HD_;
  const __bf16* Vbase = Vr + ((size_t)b * NKV_ + kvh) * L_ * HD_;

  // stage Q tile: 64x128 = 1024 async b128 chunks
  for (int c = tid; c < 1024; c += 128) {
    int r = c >> 4, d = (c & 15) * 8;
    async_b128(Qbase + (size_t)r * HD_ + d, &Qs[r][d]);
  }

  auto loadKV = [&](int buf, int j) {
    for (int c = tid; c < 512; c += 128) {       // V: async contiguous
      int key = c >> 4, d = (c & 15) * 8;
      async_b128(Vbase + (size_t)(j * 32 + key) * HD_ + d, &Vs[buf][key][d]);
    }
    for (int i = tid; i < 32 * 128; i += 128) {  // K: transpose scatter
      int key = i >> 7, d = i & 127;
      Kt[buf][d][key] = Kbase[(size_t)(j * 32 + key) * HD_ + d];
    }
  };

  loadKV(0, 0);
  wait_async0();
  __syncthreads();

  v16bf qa[4];
  {
    int r = w * 16 + l15;
    for (int kk = 0; kk < 4; kk++)
      for (int e = 0; e < 8; e++) {
        qa[kk][e]     = Qs[r][kk * 32 + half * 8 + e];
        qa[kk][8 + e] = Qs[r][kk * 32 + 16 + half * 8 + e];
      }
  }

  v8f o[8];
  for (int f = 0; f < 8; f++)
    for (int e = 0; e < 8; e++) o[f][e] = 0.0f;
  float m_i[8], l_i[8];
  for (int v = 0; v < 8; v++) { m_i[v] = -1e30f; l_i[v] = 0.0f; }

  int jmax = qt * 2 + 1;
  for (int j = 0; j <= jmax; j++) {
    const int cur = j & 1;
    if (j < jmax) loadKV(cur ^ 1, j + 1);  // stream next K/V tile

    v8f s0, s1;
    for (int e = 0; e < 8; e++) { s0[e] = 0.0f; s1[e] = 0.0f; }
    for (int kk = 0; kk < 4; kk++) {
      v16bf b0, b1;
      for (int e = 0; e < 8; e++) {
        b0[e]     = Kt[cur][kk * 32 + half * 8 + e][l15];
        b0[8 + e] = Kt[cur][kk * 32 + 16 + half * 8 + e][l15];
        b1[e]     = Kt[cur][kk * 32 + half * 8 + e][16 + l15];
        b1[8 + e] = Kt[cur][kk * 32 + 16 + half * 8 + e][16 + l15];
      }
      s0 = __builtin_amdgcn_wmma_f32_16x16x32_bf16(false, qa[kk], false, b0,
                                                   (short)0, s0, false, false);
      s1 = __builtin_amdgcn_wmma_f32_16x16x32_bf16(false, qa[kk], false, b1,
                                                   (short)0, s1, false, false);
    }

    int key0 = j * 32 + l15, key1 = key0 + 16;
    float p0[8], p1[8];
    for (int v = 0; v < 8; v++) {
      int qrow = qt * 64 + w * 16 + v + 8 * half;
      float a0 = (key0 <= qrow) ? s0[v] * SCALE_ : -1e30f;
      float a1 = (key1 <= qrow) ? s1[v] * SCALE_ : -1e30f;
      float t = fmaxf(a0, a1);
      for (int msk = 1; msk < 16; msk <<= 1) t = fmaxf(t, __shfl_xor(t, msk, 32));
      float mn   = fmaxf(m_i[v], t);
      float corr = __expf(m_i[v] - mn);
      float e0 = __expf(a0 - mn);
      float e1 = __expf(a1 - mn);
      float rs = e0 + e1;
      for (int msk = 1; msk < 16; msk <<= 1) rs += __shfl_xor(rs, msk, 32);
      l_i[v] = l_i[v] * corr + rs;
      m_i[v] = mn;
      for (int f = 0; f < 8; f++) o[f][v] *= corr;
      p0[v] = e0; p1[v] = e1;
    }

    for (int v = 0; v < 8; v++) {
      Ps[w][v + 8 * half][l15]      = (__bf16)p0[v];
      Ps[w][v + 8 * half][16 + l15] = (__bf16)p1[v];
    }
    __builtin_amdgcn_wave_barrier();
    v16bf ap;
    for (int e = 0; e < 8; e++) {
      ap[e]     = Ps[w][l15][half * 8 + e];
      ap[8 + e] = Ps[w][l15][16 + half * 8 + e];
    }
    for (int f = 0; f < 8; f++) {
      v16bf bv;
      for (int e = 0; e < 8; e++) {
        bv[e]     = Vs[cur][half * 8 + e][f * 16 + l15];
        bv[8 + e] = Vs[cur][16 + half * 8 + e][f * 16 + l15];
      }
      o[f] = __builtin_amdgcn_wmma_f32_16x16x32_bf16(false, ap, false, bv,
                                                     (short)0, o[f], false, false);
    }

    wait_async0();
    __syncthreads();
  }

  for (int v = 0; v < 8; v++) {
    int pos = qt * 64 + w * 16 + v + 8 * half;
    float inv = 1.0f / l_i[v];
    size_t ob = ((size_t)b * L_ + pos) * D_ + h * HD_;
    for (int f = 0; f < 8; f++)
      Oa[ob + f * 16 + l15] = (__bf16)(o[f][v] * inv);
  }
}

// ---------------------------------------------------------------- launch
extern "C" void kernel_launch(void* const* d_in, const int* in_sizes, int n_in,
                              void* d_out, int out_size, void* d_ws, size_t ws_size,
                              hipStream_t stream) {
  const float* x  = (const float*)d_in[0];
  const float* Wq = (const float*)d_in[1];
  const float* Wk = (const float*)d_in[2];
  const float* Wv = (const float*)d_in[3];
  const float* Wo = (const float*)d_in[4];

  char* ws = (char*)d_ws;
  size_t off = 0;
  auto alloc = [&](size_t bytes) -> char* {
    char* p = ws + off;
    off += (bytes + 255) & ~(size_t)255;
    return p;
  };

  const size_t N_XB   = 4096ull * 2048;
  const size_t N_WQKV = 4096ull * 2048;
  const size_t N_WOB  = 2048ull * 2048;
  const size_t N_QKV  = 4096ull * 4096;
  const size_t N_QR   = 2ull * NH_  * L_ * HD_;
  const size_t N_KR   = 2ull * NKV_ * L_ * HD_;

  __bf16* xb    = (__bf16*)alloc(N_XB * 2);
  __bf16* wqkv  = (__bf16*)alloc(N_WQKV * 2);
  __bf16* wob   = (__bf16*)alloc(N_WOB * 2);
  __bf16* qkv   = (__bf16*)alloc(N_QKV * 2);
  __bf16* qr    = (__bf16*)alloc(N_QR * 2);
  __bf16* kr    = (__bf16*)alloc(N_KR * 2);
  __bf16* vr    = (__bf16*)alloc(N_KR * 2);
  __bf16* oattn = qkv;  // qkv dead after rope_split; reuse

  cvt_f32_bf16<<<(int)(N_XB / 256), 256, 0, stream>>>(x, xb, (int)N_XB);
  cvt_f32_bf16<<<4194304 / 256, 256, 0, stream>>>(Wq, wqkv,           4194304);
  cvt_f32_bf16<<<2097152 / 256, 256, 0, stream>>>(Wk, wqkv + 4194304, 2097152);
  cvt_f32_bf16<<<2097152 / 256, 256, 0, stream>>>(Wv, wqkv + 6291456, 2097152);
  cvt_f32_bf16<<<4194304 / 256, 256, 0, stream>>>(Wo, wob,            4194304);

  gemm_bf16<__bf16><<<dim3(32, 64), 256, 0, stream>>>(xb, wqkv, qkv, 4096, 4096, 2048);
  rope_split<<<(int)(N_QKV / 256), 256, 0, stream>>>(qkv, qr, kr, vr);
  flash_gqa<<<B_ * NH_ * (L_ / 64), 128, 0, stream>>>(qr, kr, vr, oattn);
  gemm_bf16<float><<<dim3(16, 64), 256, 0, stream>>>(oattn, wob, (float*)d_out,
                                                     4096, 2048, 2048);
}
